// MultiHeadTemperalPoolAttentionNoTokenSpatioFirst_19035295056365
// MI455X (gfx1250) — matrix operand order repaired and checked by reading, not compiled, not run on hardware
//
#include <hip/hip_runtime.h>
#include <hip/hip_bf16.h>

typedef __attribute__((ext_vector_type(16))) _Float16 v16h;
typedef __attribute__((ext_vector_type(8)))  _Float16 v8h;
typedef __attribute__((ext_vector_type(8)))  float    v8f;

// Problem constants (fixed by the reference: t=8, w=h=14, b=8, C=512, H=8)
#define TT     8
#define WH     196           // 14*14
#define WHP    208           // wh padded to 13*16
#define KPAD   224           // wh padded to 7*32 (K-dim for P@V)
#define BATCH  8
#define CMODEL 512
#define DH     64
#define NHEAD  8
#define MROWS  12544         // 1568 tokens * 8 batch
#define SB     512           // spatial batches = H*b*t
#define TB     12544         // temporal batches = H*b*wh

__device__ __forceinline__ v8f wmma16x16x32(v16h a, v16h b, v8f c) {
  return __builtin_amdgcn_wmma_f32_16x16x32_f16(
      /*neg_a=*/false, a, /*neg_b=*/false, b,
      /*c_mod=*/(short)0, c, /*reuse_a=*/false, /*reuse_b=*/false);
}

__global__ void f32_to_f16_k(const float* __restrict__ in,
                             _Float16* __restrict__ out, int n) {
  int i = blockIdx.x * blockDim.x + threadIdx.x;
  if (i < n) out[i] = (_Float16)in[i];
}

__global__ void copy_f32_k(const float* __restrict__ in,
                           float* __restrict__ out, int n) {
  int i = blockIdx.x * blockDim.x + threadIdx.x;
  if (i < n) out[i] = in[i];
}

// C = A(MxK) * B^T(KxN) + bias, A/B f16 row-major (B given as (N,K) row-major).
// mode 0: store f16 to outH[row*N+col].
// mode 1: store f32 to outF[(row+8)*N+col]  (shift past CLS row of (1569,8,512)).
// Wave computes a 16x64 tile; block = 8 waves = 128(M) x 64(N).
__global__ void gemm_f16_wmma(const _Float16* __restrict__ A,
                              const _Float16* __restrict__ Bw,
                              const float* __restrict__ bias,
                              _Float16* __restrict__ outH,
                              float* __restrict__ outF,
                              int M, int N, int K, int mode) {
  const int lane = threadIdx.x & 31;
  const int wave = threadIdx.x >> 5;
  const int m0 = blockIdx.y * 128 + wave * 16;
  const int n0 = blockIdx.x * 64;
  const int l15 = lane & 15;
  const int kA = (lane >> 4) * 16;   // A-frag K sub-offset per half-wave
  const int kB = (lane >> 4) * 16;   // B-frag K sub-offset per half-wave

  v8f c[4] = {v8f{}, v8f{}, v8f{}, v8f{}};
  const _Float16* arow = A + (size_t)(m0 + l15) * K;

  for (int k0 = 0; k0 < K; k0 += 32) {
    v8h alo = *(const v8h*)(arow + k0 + kA);
    v8h ahi = *(const v8h*)(arow + k0 + kA + 8);
    v16h a = __builtin_shufflevector(alo, ahi, 0,1,2,3,4,5,6,7,8,9,10,11,12,13,14,15);
#pragma unroll
    for (int j = 0; j < 4; ++j) {
      v16h b = *(const v16h*)(Bw + (size_t)(n0 + j * 16 + l15) * K + k0 + kB);
      c[j] = wmma16x16x32(a, b, c[j]);
    }
  }

  const int rbase = m0 + (lane >> 4) * 8;
#pragma unroll
  for (int j = 0; j < 4; ++j) {
    int col = n0 + j * 16 + l15;
    float bv = bias[col];
#pragma unroll
    for (int r = 0; r < 8; ++r) {
      float v = c[j][r] + bv;
      size_t idx = (size_t)(rbase + r) * N + col;
      if (mode == 0) outH[idx] = (_Float16)v;
      else           outF[idx + (size_t)8 * N] = v;
    }
  }
}

// qkv (12544 x 1536) f16 -> q_s/k_s (512, 208, 64) zero-padded rows.
__global__ void rearrange_qk_k(const _Float16* __restrict__ qkv,
                               _Float16* __restrict__ q_s,
                               _Float16* __restrict__ k_s) {
  int i = blockIdx.x * blockDim.x + threadIdx.x;
  if (i >= SB * WHP * DH) return;
  int d  = i & 63;
  int p  = (i >> 6) % WHP;
  int bs = i / (WHP * DH);              // bs = Hh*64 + b*8 + ti
  int ti = bs & 7, b = (bs >> 3) & 7, Hh = bs >> 6;
  _Float16 qv = (_Float16)0.f, kv = (_Float16)0.f;
  if (p < WH) {
    size_t row = (size_t)((ti * WH + p) * BATCH + b) * (3 * CMODEL);
    int col = Hh * DH + d;
    qv = qkv[row + col];
    kv = qkv[row + CMODEL + col];
  }
  q_s[i] = qv;
  k_s[i] = kv;
}

// qkv v-section -> v_t (512, 64, 224): transposed + zero-padded keys.
__global__ void rearrange_v_k(const _Float16* __restrict__ qkv,
                              _Float16* __restrict__ v_t) {
  int i = blockIdx.x * blockDim.x + threadIdx.x;
  if (i >= SB * DH * KPAD) return;
  int p  = i % KPAD;
  int d  = (i / KPAD) & 63;
  int bs = i / (KPAD * DH);
  int ti = bs & 7, b = (bs >> 3) & 7, Hh = bs >> 6;
  _Float16 v = (_Float16)0.f;
  if (p < WH)
    v = qkv[(size_t)((ti * WH + p) * BATCH + b) * (3 * CMODEL) + 2 * CMODEL + Hh * DH + d];
  v_t[i] = v;
}

// Spatial SDPA. grid = (13 q-tiles, 512 batches), block = 32 (one wave).
// Writes attn probs (f32, 512x196x196) and out_s (f16, 512x196x64).
__global__ void spatial_attn_k(const _Float16* __restrict__ q_s,
                               const _Float16* __restrict__ k_s,
                               const _Float16* __restrict__ v_t,
                               _Float16* __restrict__ out_s,
                               float* __restrict__ attn) {
  __shared__ float    sc[16 * WHP];    // scores -> probs (f32)
  __shared__ _Float16 pb[16 * KPAD];   // probs as f16, zero-padded K

  const int lane = threadIdx.x & 31;
  const int qt = blockIdx.x;
  const int bs = blockIdx.y;
  const int l15 = lane & 15;
  const int kA = (lane >> 4) * 16;
  const int kB = (lane >> 4) * 16;

  const _Float16* Q  = q_s + (size_t)bs * WHP * DH;
  const _Float16* Kp = k_s + (size_t)bs * WHP * DH;
  const _Float16* VT = v_t + (size_t)bs * DH * KPAD;

  const _Float16* qrow = Q + (size_t)(qt * 16 + l15) * DH;

  // scores = Q * K^T / sqrt(dh): 13 key tiles, K-dim = 64 (2 WMMA steps)
  for (int j = 0; j < 13; ++j) {
    v8f c = {};
#pragma unroll
    for (int k0 = 0; k0 < DH; k0 += 32) {
      v8h alo = *(const v8h*)(qrow + k0 + kA);
      v8h ahi = *(const v8h*)(qrow + k0 + kA + 8);
      v16h a = __builtin_shufflevector(alo, ahi, 0,1,2,3,4,5,6,7,8,9,10,11,12,13,14,15);
      v16h b = *(const v16h*)(Kp + (size_t)(j * 16 + l15) * DH + k0 + kB);
      c = wmma16x16x32(a, b, c);
    }
#pragma unroll
    for (int r = 0; r < 8; ++r) {
      int row = (lane >> 4) * 8 + r;
      sc[row * WHP + j * 16 + l15] = c[r] * 0.125f;
    }
  }
  __syncthreads();

  // row softmax over 196 valid keys (lanes 0..15, one row each)
  if (lane < 16) {
    float m = -1e30f;
    for (int ccol = 0; ccol < WH; ++ccol) m = fmaxf(m, sc[lane * WHP + ccol]);
    float s = 0.f;
    for (int ccol = 0; ccol < WH; ++ccol) {
      float e = __expf(sc[lane * WHP + ccol] - m);
      sc[lane * WHP + ccol] = e;
      s += e;
    }
    float inv = 1.f / s;
    for (int ccol = 0; ccol < WH; ++ccol) sc[lane * WHP + ccol] *= inv;
  }
  __syncthreads();

  // write attn output (f32) and pack probs to f16 LDS (zero pad K to 224)
  for (int idx = lane; idx < 16 * WHP; idx += 32) {
    int row = idx / WHP, col = idx % WHP;
    int qrowg = qt * 16 + row;
    if (qrowg < WH && col < WH)
      attn[(size_t)bs * WH * WH + (size_t)qrowg * WH + col] = sc[idx];
  }
  for (int idx = lane; idx < 16 * KPAD; idx += 32) {
    int row = idx / KPAD, col = idx % KPAD;
    pb[idx] = (col < WH) ? (_Float16)sc[row * WHP + col] : (_Float16)0.f;
  }
  __syncthreads();

  // out = P (16x224) @ V (224x64), B-frags contiguous from transposed V
  v8f c2[4] = {v8f{}, v8f{}, v8f{}, v8f{}};
  for (int kk = 0; kk < KPAD; kk += 32) {
    v8h plo = *(const v8h*)(&pb[l15 * KPAD + kk + kA]);
    v8h phi = *(const v8h*)(&pb[l15 * KPAD + kk + kA + 8]);
    v16h a = __builtin_shufflevector(plo, phi, 0,1,2,3,4,5,6,7,8,9,10,11,12,13,14,15);
#pragma unroll
    for (int j = 0; j < 4; ++j) {
      v16h b = *(const v16h*)(VT + (size_t)(j * 16 + l15) * KPAD + kk + kB);
      c2[j] = wmma16x16x32(a, b, c2[j]);
    }
  }
#pragma unroll
  for (int j = 0; j < 4; ++j) {
#pragma unroll
    for (int r = 0; r < 8; ++r) {
      int row = qt * 16 + (lane >> 4) * 8 + r;
      if (row < WH)
        out_s[(size_t)bs * WH * DH + (size_t)row * DH + j * 16 + l15] = (_Float16)c2[j][r];
    }
  }
}

// Temporal SDPA over t=8 (tiny -> VALU). block = 256 = 4 groups of 64.
__global__ void temporal_attn_k(const _Float16* __restrict__ q_s,
                                const _Float16* __restrict__ k_s,
                                const _Float16* __restrict__ out_s,
                                _Float16* __restrict__ yb) {
  __shared__ float sc[4][TT][TT];
  const int g = threadIdx.x & 63;
  const int grp = threadIdx.x >> 6;
  const int bt = blockIdx.x * 4 + grp;      // (Hh*8+b)*196 + p
  const int p = bt % WH;
  const int hb = bt / WH;                   // Hh*8 + b
  const int Hh = hb >> 3, b = hb & 7;
  const size_t qbase = (size_t)hb * TT * WHP * DH + (size_t)p * DH;
  const size_t obase = (size_t)hb * TT * WH * DH + (size_t)p * DH;

  {
    int ti = g >> 3, tj = g & 7;
    const _Float16* qr = q_s + qbase + (size_t)ti * WHP * DH;
    const _Float16* kr = k_s + qbase + (size_t)tj * WHP * DH;
    float s = 0.f;
#pragma unroll
    for (int d = 0; d < DH; ++d) s += (float)qr[d] * (float)kr[d];
    sc[grp][ti][tj] = s * 0.125f;
  }
  __syncthreads();
  if (g < TT) {
    float m = -1e30f;
#pragma unroll
    for (int j = 0; j < TT; ++j) m = fmaxf(m, sc[grp][g][j]);
    float s = 0.f;
#pragma unroll
    for (int j = 0; j < TT; ++j) { float e = __expf(sc[grp][g][j] - m); sc[grp][g][j] = e; s += e; }
    float inv = 1.f / s;
#pragma unroll
    for (int j = 0; j < TT; ++j) sc[grp][g][j] *= inv;
  }
  __syncthreads();

  const int d = g;  // 0..63
  float o[TT] = {0.f, 0.f, 0.f, 0.f, 0.f, 0.f, 0.f, 0.f};
#pragma unroll
  for (int tj = 0; tj < TT; ++tj) {
    float v = (float)out_s[obase + (size_t)tj * WH * DH + d];
#pragma unroll
    for (int ti = 0; ti < TT; ++ti) o[ti] += sc[grp][ti][tj] * v;
  }
#pragma unroll
  for (int ti = 0; ti < TT; ++ti) {
    size_t row = (size_t)((ti * WH + p) * BATCH + b);
    yb[row * CMODEL + Hh * DH + d] = (_Float16)o[ti];
  }
}

extern "C" void kernel_launch(void* const* d_in, const int* in_sizes, int n_in,
                              void* d_out, int out_size, void* d_ws, size_t ws_size,
                              hipStream_t stream) {
  const float* q    = (const float*)d_in[0];
  // d_in[1] (k) and d_in[2] (value) are unused by the reference math.
  const float* Wqkv = (const float*)d_in[3];  // (1536, 512)
  const float* bqkv = (const float*)d_in[4];  // (1536,)
  const float* Wout = (const float*)d_in[5];  // (512, 512)
  const float* bout = (const float*)d_in[6];  // (512,)
  float* out = (float*)d_out;

  // workspace carve-up (f16 elements)
  _Float16* ws = (_Float16*)d_ws;
  size_t off = 0;
  _Float16* xb    = ws + off; off += (size_t)MROWS * CMODEL;       // 12544x512
  _Float16* wqkvb = ws + off; off += (size_t)3 * CMODEL * CMODEL;  // 1536x512
  _Float16* woutb = ws + off; off += (size_t)CMODEL * CMODEL;      // 512x512
  _Float16* qkvb  = ws + off; off += (size_t)MROWS * 3 * CMODEL;   // 12544x1536
  _Float16* q_s   = ws + off; off += (size_t)SB * WHP * DH;        // 512x208x64
  _Float16* k_s   = ws + off; off += (size_t)SB * WHP * DH;
  _Float16* v_t   = ws + off; off += (size_t)SB * DH * KPAD;       // 512x64x224
  _Float16* outs  = ws + off; off += (size_t)SB * WH * DH;         // 512x196x64
  _Float16* yb    = ws + off; off += (size_t)MROWS * CMODEL;

  float* attn = out + (size_t)1569 * BATCH * CMODEL;  // 2nd output region

  const int nx = MROWS * CMODEL;
  f32_to_f16_k<<<(nx + 255) / 256, 256, 0, stream>>>(q + BATCH * CMODEL, xb, nx);
  f32_to_f16_k<<<(3 * CMODEL * CMODEL + 255) / 256, 256, 0, stream>>>(Wqkv, wqkvb, 3 * CMODEL * CMODEL);
  f32_to_f16_k<<<(CMODEL * CMODEL + 255) / 256, 256, 0, stream>>>(Wout, woutb, CMODEL * CMODEL);

  // QKV projection: (12544x512) @ (512x1536) + bias
  gemm_f16_wmma<<<dim3((3 * CMODEL) / 64, MROWS / 128), 256, 0, stream>>>(
      xb, wqkvb, bqkv, qkvb, nullptr, MROWS, 3 * CMODEL, CMODEL, 0);

  rearrange_qk_k<<<(SB * WHP * DH + 255) / 256, 256, 0, stream>>>(qkvb, q_s, k_s);
  rearrange_v_k<<<(SB * DH * KPAD + 255) / 256, 256, 0, stream>>>(qkvb, v_t);

  spatial_attn_k<<<dim3(13, SB), 32, 0, stream>>>(q_s, k_s, v_t, outs, attn);

  temporal_attn_k<<<TB / 4, 256, 0, stream>>>(q_s, k_s, outs, yb);

  // Output projection: (12544x512) @ (512x512) + bias -> f32 rows 8.. of d_out
  gemm_f16_wmma<<<dim3(CMODEL / 64, MROWS / 128), 256, 0, stream>>>(
      yb, woutb, bout, nullptr, out, MROWS, CMODEL, CMODEL, 1);

  // CLS row passthrough: d_out[0:4096] = q[0, :, :]
  copy_f32_k<<<16, 256, 0, stream>>>(q, out, BATCH * CMODEL);
}
